// MultiheadAttention_88235808129319
// MI455X (gfx1250) — compile-verified, matrix operand order checked by reading, not run on hardware
//
#include <hip/hip_runtime.h>

// ---------------- problem constants (from reference) ----------------
#define NE 8            // experts
#define DIN 1024        // D
#define DH 1024         // H
#define NT 8192         // T = B*L = 4*2048
#define ROWT 32         // rows (gathered tokens) per block
#define MTILES (NT / ROWT)   // 256 worst-case row tiles per expert

typedef __attribute__((ext_vector_type(16))) __bf16 bf16x16;
typedef __attribute__((ext_vector_type(8)))  float  f32x8;

union Frag { unsigned int u[8]; bf16x16 v; };

// ---------------- helpers ----------------
__device__ __forceinline__ unsigned short f2bf(float f) {
    unsigned int u = __float_as_uint(f);
    unsigned int r = u + 0x7FFFu + ((u >> 16) & 1u);   // round-to-nearest-even
    return (unsigned short)(r >> 16);
}

// branch-free tanh: (e^{2x}-1)/(e^{2x}+1) via v_exp_f32 + v_rcp_f32
__device__ __forceinline__ float fast_tanh(float x) {
    float xc = fminf(fmaxf(x, -8.0f), 8.0f);
    float e2 = __builtin_amdgcn_exp2f(xc * 2.885390081777927f);   // e^{2x}
    return (e2 - 1.0f) * __builtin_amdgcn_rcpf(e2 + 1.0f);
}

__device__ __forceinline__ float gelu_tanh(float x) {
    const float c = 0.7978845608028654f;               // sqrt(2/pi)
    float x3 = x * x * x;
    return 0.5f * x * (1.0f + fast_tanh(c * (x + 0.044715f * x3)));
}

// load one 16-bit WMMA fragment (ISA 16x32 layout: halves split at lane 16,
// packed K pairs) -> two b128 loads
__device__ __forceinline__ void ld8(Frag& f, const unsigned int* __restrict__ p,
                                    long long ro, int half) {
#pragma unroll
    for (int i = 0; i < 4; ++i) f.u[i]     = p[ro + half * 4 + i];
#pragma unroll
    for (int i = 0; i < 4; ++i) f.u[4 + i] = p[ro + 8 + half * 4 + i];
}

// ---------------- utility kernels ----------------
__global__ void zero_f32v4(float4* p, int n4) {
    int i = blockIdx.x * blockDim.x + threadIdx.x;
    if (i < n4) p[i] = make_float4(0.f, 0.f, 0.f, 0.f);
}

__global__ void zero_f32(float* p, int n) {
    int i = blockIdx.x * blockDim.x + threadIdx.x;
    if (i < n) p[i] = 0.0f;
}

__global__ void cvt_x_bf16(const float4* x, uint2* xbf, int n4) {
    int i = blockIdx.x * blockDim.x + threadIdx.x;
    if (i >= n4) return;
    float4 v = x[i];
    uint2 o;
    o.x = (unsigned)f2bf(v.x) | ((unsigned)f2bf(v.y) << 16);
    o.y = (unsigned)f2bf(v.z) | ((unsigned)f2bf(v.w) << 16);
    xbf[i] = o;
}

// out[e][b][a] = in[e][a][b] with in dims [NE][d0][d1]; also fp32 -> bf16
__global__ void cvt_transpose_bf16(const float* in, unsigned short* out, int d0, int d1) {
    long long i = (long long)blockIdx.x * blockDim.x + threadIdx.x;
    long long per = (long long)d0 * d1;
    if (i >= (long long)NE * per) return;
    int e = (int)(i / per);
    long long r = i % per;
    int a = (int)(r / d1);
    int b = (int)(r % d1);
    out[(long long)e * per + (long long)b * d0 + a] = f2bf(in[i]);
}

// ---------------- gating: one wave32 per token ----------------
__global__ __launch_bounds__(256) void gate_kernel(
    const float* __restrict__ x, const float* __restrict__ wg,
    int* lists, float* gatev, int* cnt,
    float* gsum, float* psum, float* zsum)
{
    const float LOG2E = 1.4426950408889634f;
    const float LN2   = 0.6931471805599453f;
    int wave = threadIdx.x >> 5;
    int lane = threadIdx.x & 31;
    int t = blockIdx.x * 8 + wave;
    if (t >= NT) return;

    float acc[NE];
#pragma unroll
    for (int e = 0; e < NE; ++e) acc[e] = 0.0f;

    const float* xr = x + (long long)t * DIN;
    for (int d = lane; d < DIN; d += 32) {
        float xv = xr[d];
#pragma unroll
        for (int e = 0; e < NE; ++e) acc[e] += xv * wg[d * NE + e];
    }
#pragma unroll
    for (int e = 0; e < NE; ++e)
        for (int m = 16; m >= 1; m >>= 1) acc[e] += __shfl_xor(acc[e], m, 32);

    if (lane == 0) {
        float mx = acc[0];
#pragma unroll
        for (int e = 1; e < NE; ++e) mx = fmaxf(mx, acc[e]);
        float ex[NE], s = 0.0f;
#pragma unroll
        for (int e = 0; e < NE; ++e) {
            ex[e] = __builtin_amdgcn_exp2f((acc[e] - mx) * LOG2E);
            s += ex[e];
        }
        float inv = __builtin_amdgcn_rcpf(s);
        float lse = mx + __builtin_amdgcn_logf(s) * LN2;
        float p[NE];
#pragma unroll
        for (int e = 0; e < NE; ++e) { p[e] = ex[e] * inv; atomicAdd(&psum[e], p[e]); }

        int e0 = 0;
#pragma unroll
        for (int e = 1; e < NE; ++e) if (p[e] > p[e0]) e0 = e;
        int e1 = (e0 == 0) ? 1 : 0;
#pragma unroll
        for (int e = 0; e < NE; ++e) if (e != e0 && p[e] > p[e1]) e1 = e;

        float g0 = p[e0], g1 = p[e1];
        atomicAdd(&gsum[e0], g0);
        atomicAdd(&gsum[e1], g1);
        int p0 = atomicAdd(&cnt[e0], 1);
        lists[e0 * NT + p0] = t; gatev[e0 * NT + p0] = g0;
        int p1 = atomicAdd(&cnt[e1], 1);
        lists[e1 * NT + p1] = t; gatev[e1 * NT + p1] = g1;
        atomicAdd(zsum, lse * lse);
    }
}

// ---------------- fused expert tile ----------------
// block = 256 threads = 8 waves; 32 gathered rows of one expert per block.
// wave w owns output columns [w*128, w*128+128); A fragments come straight
// from global (K-contiguous), B fragments are double-buffered across the
// 8 column sub-tiles so loads run ahead of the wmma stream.
__global__ __launch_bounds__(256) void expert_kernel(
    const unsigned short* __restrict__ xbf,     // [NT][DIN] bf16
    const unsigned short* __restrict__ wInT,    // [NE][DH][DIN] bf16 (K contiguous)
    const unsigned short* __restrict__ wOutT,   // [NE][DIN][DH] bf16 (K contiguous)
    const int* __restrict__ lists, const float* __restrict__ gatev,
    const int* __restrict__ cnt, float* y)
{
    __shared__ unsigned short hidden[ROWT * DH];   // 64 KB bf16 hidden tile

    int e    = blockIdx.x >> 8;        // / MTILES(256)
    int tile = blockIdx.x & 255;
    int c = cnt[e];
    int m0 = tile * ROWT;
    if (m0 >= c) return;               // uniform per block

    int tid  = threadIdx.x;
    int wave = tid >> 5, lane = tid & 31;
    int half = lane >> 4, lan15 = lane & 15;
    int lbase = e * NT + m0;

    // tokens feeding this lane's A fragments (row = lan15 of each 16-row sub-tile)
    int gr0 = m0 + lan15, gr1 = m0 + 16 + lan15;
    int tokA0 = (gr0 < c) ? lists[lbase + lan15] : 0;
    int tokA1 = (gr1 < c) ? lists[lbase + 16 + lan15] : 0;
    long long xr0 = (long long)tokA0 * (DIN / 2);
    long long xr1 = (long long)tokA1 * (DIN / 2);

    const unsigned int* xbf32 = (const unsigned int*)xbf;
    const unsigned int* h32   = (const unsigned int*)hidden;
    const unsigned int* wi32  = (const unsigned int*)(wInT  + (long long)e * DH * DIN);
    const unsigned int* wo32  = (const unsigned int*)(wOutT + (long long)e * DIN * DH);

    // ---------- GEMM1: 32 x DH, K = DIN ----------
    f32x8 acc[2][8];
#pragma unroll
    for (int mt = 0; mt < 2; ++mt)
#pragma unroll
        for (int j = 0; j < 8; ++j)
#pragma unroll
            for (int i = 0; i < 8; ++i) acc[mt][j][i] = 0.0f;

    for (int kb2 = 0; kb2 < DIN / 2; kb2 += 16) {    // 16 u32 = 32 bf16 of K
        Frag a0, a1;
        ld8(a0, xbf32, xr0 + kb2, half);
        ld8(a1, xbf32, xr1 + kb2, half);
        Frag b[2];
        ld8(b[0], wi32, (long long)(wave * 128 + lan15) * (DIN / 2) + kb2, half);
#pragma unroll
        for (int j = 0; j < 8; ++j) {
            if (j < 7) {
                long long ro = (long long)(wave * 128 + (j + 1) * 16 + lan15) * (DIN / 2) + kb2;
                ld8(b[(j + 1) & 1], wi32, ro, half);
                __builtin_prefetch((const void*)(wi32 + ro + 16), 0, 1); // next K slab
            }
            acc[0][j] = __builtin_amdgcn_wmma_f32_16x16x32_bf16(
                false, a0.v, false, b[j & 1].v, (short)0, acc[0][j], false, false);
            acc[1][j] = __builtin_amdgcn_wmma_f32_16x16x32_bf16(
                false, a1.v, false, b[j & 1].v, (short)0, acc[1][j], false, false);
        }
    }

    // gelu + pack to bf16 hidden tile (C/D layout: vgpr i -> row i + 8*half, col = lan15)
#pragma unroll
    for (int mt = 0; mt < 2; ++mt)
#pragma unroll
        for (int j = 0; j < 8; ++j)
#pragma unroll
            for (int i = 0; i < 8; ++i) {
                int m = mt * 16 + i + 8 * half;
                hidden[m * DH + wave * 128 + j * 16 + lan15] = f2bf(gelu_tanh(acc[mt][j][i]));
            }
    __syncthreads();   // the only barrier: hidden is read across all waves

    // ---------- GEMM2: 32 x DIN, K = DH ----------
    f32x8 acc2[2][8];
#pragma unroll
    for (int mt = 0; mt < 2; ++mt)
#pragma unroll
        for (int j = 0; j < 8; ++j)
#pragma unroll
            for (int i = 0; i < 8; ++i) acc2[mt][j][i] = 0.0f;

    for (int kb2 = 0; kb2 < DH / 2; kb2 += 16) {
        Frag a0, a1;
        ld8(a0, h32, (long long)lan15 * (DH / 2) + kb2, half);
        ld8(a1, h32, (long long)(16 + lan15) * (DH / 2) + kb2, half);
        Frag b[2];
        ld8(b[0], wo32, (long long)(wave * 128 + lan15) * (DH / 2) + kb2, half);
#pragma unroll
        for (int j = 0; j < 8; ++j) {
            if (j < 7) {
                long long ro = (long long)(wave * 128 + (j + 1) * 16 + lan15) * (DH / 2) + kb2;
                ld8(b[(j + 1) & 1], wo32, ro, half);
                __builtin_prefetch((const void*)(wo32 + ro + 16), 0, 1);
            }
            acc2[0][j] = __builtin_amdgcn_wmma_f32_16x16x32_bf16(
                false, a0.v, false, b[j & 1].v, (short)0, acc2[0][j], false, false);
            acc2[1][j] = __builtin_amdgcn_wmma_f32_16x16x32_bf16(
                false, a1.v, false, b[j & 1].v, (short)0, acc2[1][j], false, false);
        }
    }

    // gate/token info for this lane's output rows (row = i + 8*half per sub-tile)
    int   tokR[2][8];
    float gR[2][8];
#pragma unroll
    for (int mt = 0; mt < 2; ++mt)
#pragma unroll
        for (int i = 0; i < 8; ++i) {
            int gr = m0 + mt * 16 + i + 8 * half;
            bool ok = gr < c;
            tokR[mt][i] = ok ? lists[e * NT + gr] : 0;
            gR[mt][i]   = ok ? gatev[e * NT + gr] : 0.0f;
        }

    // scale by gate and scatter-add into y (two experts per token -> atomics)
#pragma unroll
    for (int mt = 0; mt < 2; ++mt)
#pragma unroll
        for (int j = 0; j < 8; ++j)
#pragma unroll
            for (int i = 0; i < 8; ++i) {
                float v = gR[mt][i] * acc2[mt][j][i];
                atomicAdd(&y[(long long)tokR[mt][i] * DIN + wave * 128 + j * 16 + lan15], v);
            }
}

// ---------------- aux losses ----------------
__global__ void loss_kernel(const float* gsum, const float* psum, const int* cnt,
                            const float* zsum, float* out_loss)
{
    if (threadIdx.x != 0 || blockIdx.x != 0) return;
    float s = 0.0f;
    for (int e = 0; e < NE; ++e) s += fabsf(gsum[e]);
    s = fmaxf(s, 1e-12f);
    float v[NE], mean = 0.0f;
    for (int e = 0; e < NE; ++e) { v[e] = gsum[e] / s; mean += v[e]; }
    mean /= NE;
    float var = 0.0f;
    for (int e = 0; e < NE; ++e) { float d = v[e] - mean; var += d * d; }
    var /= (NE - 1);                                  // unbiased, like torch .var()
    float cv = var / (mean * mean + 1e-10f);

    float sp = 0.0f, sc = 0.0f;
    for (int e = 0; e < NE; ++e) { sp += fabsf(psum[e]); sc += fabsf((float)cnt[e]); }
    sp = fmaxf(sp, 1e-12f); sc = fmaxf(sc, 1e-12f);
    float sw = 0.0f;
    for (int e = 0; e < NE; ++e) sw += (psum[e] / sp) * ((float)cnt[e] / sc);
    sw *= (float)NE;

    float zl = zsum[0] / (float)NT;
    out_loss[0] = 0.01f * cv + 0.01f * sw + 0.001f * zl;
}

// ---------------- launch ----------------
extern "C" void kernel_launch(void* const* d_in, const int* in_sizes, int n_in,
                              void* d_out, int out_size, void* d_ws, size_t ws_size,
                              hipStream_t stream)
{
    (void)in_sizes; (void)n_in; (void)out_size; (void)ws_size;
    const float* x     = (const float*)d_in[0];   // [T, D]
    const float* wgate = (const float*)d_in[1];   // [D, E]
    const float* w_in  = (const float*)d_in[2];   // [E, D, H]
    const float* w_out = (const float*)d_in[3];   // [E, H, D]

    float* y    = (float*)d_out;                  // [T, D]
    float* loss = (float*)d_out + (long long)NT * DIN;

    // workspace layout
    char* ws = (char*)d_ws;
    const long long XD  = (long long)NT * DIN;
    const long long WSZ = (long long)NE * DH * DIN;
    unsigned short* xbf   = (unsigned short*)ws;   ws += XD * 2;
    unsigned short* wInT  = (unsigned short*)ws;   ws += WSZ * 2;
    unsigned short* wOutT = (unsigned short*)ws;   ws += WSZ * 2;
    int*   lists = (int*)ws;                       ws += (long long)NE * NT * 4;
    float* gatev = (float*)ws;                     ws += (long long)NE * NT * 4;
    int*   cnt   = (int*)ws;                       ws += NE * 4;
    float* gsum  = (float*)ws;                     ws += NE * 4;
    float* psum  = (float*)ws;                     ws += NE * 4;
    float* zsum  = (float*)ws;

    const int TPB = 256;
    // 1) zero output accumulator + routing stats (ws is not re-zeroed between replays)
    zero_f32v4<<<(int)((XD / 4 + TPB - 1) / TPB), TPB, 0, stream>>>((float4*)y, (int)(XD / 4));
    zero_f32<<<1, 32, 0, stream>>>((float*)cnt, NE * 3 + 1);   // cnt|gsum|psum|zsum contiguous

    // 2) precision conversion / weight transposes (K-contiguous B operands)
    cvt_x_bf16<<<(int)((XD / 4 + TPB - 1) / TPB), TPB, 0, stream>>>(
        (const float4*)x, (uint2*)xbf, (int)(XD / 4));
    cvt_transpose_bf16<<<(int)((WSZ + TPB - 1) / TPB), TPB, 0, stream>>>(w_in,  wInT,  DIN, DH);
    cvt_transpose_bf16<<<(int)((WSZ + TPB - 1) / TPB), TPB, 0, stream>>>(w_out, wOutT, DH, DIN);

    // 3) gating + routing lists
    gate_kernel<<<NT / 8, TPB, 0, stream>>>(x, wgate, lists, gatev, cnt, gsum, psum, zsum);

    // 4) fused grouped expert GEMMs (bf16 WMMA, 32-row tiles)
    expert_kernel<<<NE * MTILES, TPB, 0, stream>>>(xbf, wInT, wOutT, lists, gatev, cnt, y);

    // 5) aux loss scalar
    loss_kernel<<<1, 32, 0, stream>>>(gsum, psum, cnt, zsum, loss);
}
